// SpatialAttention_52089363366010
// MI455X (gfx1250) — matrix operand order repaired and checked by reading, not compiled, not run on hardware
//
#include <hip/hip_runtime.h>
#include <math.h>

// Native clang ext vectors (HIP's float4 is a class; builtins need real vectors)
typedef float        f4    __attribute__((ext_vector_type(4)));
typedef float        v8f   __attribute__((ext_vector_type(8)));
typedef _Float16     v16h  __attribute__((ext_vector_type(16)));
typedef unsigned int u32x4 __attribute__((ext_vector_type(4)));
typedef int          i32x4 __attribute__((ext_vector_type(4)));
typedef int          i32x8 __attribute__((ext_vector_type(8)));

#define B_      32
#define C_      256
#define HW_     4096   // 64*64
#define Wdim    64
#define NPIX    (B_*HW_)          // 131072
#define KTAPS   98                // 2*7*7
#define LPITCH  72                // LDS row pitch (64 data + 8 pad DWORDs)
#define LCH     512               // LDS floats per channel tile (7*72=504, padded)

// ---------------------------------------------------------------------------
// Kernel 1: channel mean + max.  One thread per float4 of a spatial plane.
// Coalesced: a wave reads 512 contiguous bytes per channel step. x stays in L2
// (134 MB < 192 MB) for kernel 3's second pass.
// ---------------------------------------------------------------------------
__global__ void sa_reduce(const float* __restrict__ x,
                          float* __restrict__ avg_out,
                          float* __restrict__ max_out) {
    const int t  = blockIdx.x * blockDim.x + threadIdx.x;   // 0..32767
    const int b  = t >> 10;                                  // 1024 f4 / plane
    const int p  = t & 1023;
    const f4* xp = (const f4*)x + (size_t)b * (C_ * 1024) + p;

    f4 s = (f4)0.0f;
    f4 m = (f4)(-3.402823466e+38f);
#pragma unroll 4
    for (int c = 0; c < C_; ++c) {
        f4 v = xp[(size_t)c * 1024];
        s += v;
        m.x = fmaxf(m.x, v.x); m.y = fmaxf(m.y, v.y);
        m.z = fmaxf(m.z, v.z); m.w = fmaxf(m.w, v.w);
    }
    ((f4*)avg_out)[(size_t)b * 1024 + p] = s * (1.0f / 256.0f);
    ((f4*)max_out)[(size_t)b * 1024 + p] = m;
}

// ---------------------------------------------------------------------------
// Kernel 2: 7x7 conv (pad 3) + sigmoid.
// One block (4 waves) per output row (b,h).  The 7x64x2 input window is
// staged into pre-zeroed LDS by the Tensor Data Mover (tensor_load_to_lds,
// one descriptor per channel plane; pad_enable gives the 72-float row pitch,
// deposit at column +3 for the horizontal halo, TDM OOB-zero for the bottom
// rows, start-row offset for the top rows).  Each wave then computes 16
// output pixels as D = A(16x32 im2col,f16) x B(32x16 broadcast weights,f16)
// accumulated over 4 K-chunks with v_wmma_f32_16x16x32_f16.
// ---------------------------------------------------------------------------
__global__ void sa_conv_wmma(const float* __restrict__ avg_in,
                             const float* __restrict__ max_in,
                             const float* __restrict__ wts,   // 98 floats OIHW
                             float* __restrict__ y) {
    __shared__ float tile[2 * LCH];          // [ch][7 rows x 72 pitch]

    const int tid = threadIdx.x;

    // Zero LDS (halo columns + clipped rows must read as 0).
#pragma unroll
    for (int i = 0; i < 8; ++i) tile[tid + i * 128] = 0.0f;
    __syncthreads();

    const int b = blockIdx.x >> 6;
    const int h = blockIdx.x & 63;

    if (tid < 32) {                          // wave 0 drives the TDM
        const int r0 = (h >= 3) ? (h - 3) : 0;       // first valid tensor row
        const int rl = r0 - (h - 3);                 // first LDS tile row
        const int nrows = 7 - rl;                    // rows TDM should move
        const unsigned lds_base = (unsigned)(size_t)(&tile[0]);
        const unsigned tdim1 = (unsigned)(Wdim - r0); // rows available -> OOB 0

        // ---- D# group 1: data_size=4B, pad 8 DWORDs every 64 DWORDs ----
        i32x8 g1;
        g1[0] = (int)((2u << 16) | (1u << 20) | (5u << 22) | (7u << 25));
        g1[1] = (int)(64u << 16);            // tensor_dim0[15:0] = 64
        g1[2] = (int)(tdim1 << 16);          // tensor_dim1[15:0]
        g1[3] = (int)(64u << 16);            // tile_dim0 = 64
        g1[4] = nrows;                       // tile_dim1 (tile_dim2 = 0)
        g1[5] = 64;                          // tensor_dim0_stride = 64
        g1[6] = 0;
        g1[7] = 0;
        const i32x4 g2 = {0, 0, 0, 0};
        const i32x4 g3 = {0, 0, 0, 0};
        const i32x8 g4 = {0, 0, 0, 0, 0, 0, 0, 0};

        // ---- channel 0: avg plane ----
        {
            const unsigned long long ga =
                (unsigned long long)(size_t)(avg_in + (size_t)b * HW_ + (size_t)r0 * Wdim);
            u32x4 g0;
            g0[0] = 1u;                                       // count=1, load
            g0[1] = lds_base + (unsigned)(rl * LPITCH + 3) * 4u;
            g0[2] = (unsigned)(ga & 0xFFFFFFFFull);
            g0[3] = (unsigned)((ga >> 32) & 0x1FFFFFFull) | (2u << 30); // type=2
            __builtin_amdgcn_tensor_load_to_lds(g0, g1, g2, g3, g4, 0);
        }
        // ---- channel 1: max plane ----
        {
            const unsigned long long ga =
                (unsigned long long)(size_t)(max_in + (size_t)b * HW_ + (size_t)r0 * Wdim);
            u32x4 g0;
            g0[0] = 1u;
            g0[1] = lds_base + (unsigned)(LCH + rl * LPITCH + 3) * 4u;
            g0[2] = (unsigned)(ga & 0xFFFFFFFFull);
            g0[3] = (unsigned)((ga >> 32) & 0x1FFFFFFull) | (2u << 30);
            __builtin_amdgcn_tensor_load_to_lds(g0, g1, g2, g3, g4, 0);
        }
        __builtin_amdgcn_s_wait_tensorcnt(0);
    }
    __syncthreads();

    const int lane   = tid & 31;
    const int laneHi = lane >> 4;            // 0: lanes 0-15, 1: lanes 16-31
    const int M      = lane & 15;            // A-matrix row owned by this lane
    const int wv     = tid >> 5;             // wave in block -> w0 = wv*16
    const int wwM    = wv * 16 + M;          // output column 0..63

    v8f acc = {};
#pragma unroll
    for (int chunk = 0; chunk < 4; ++chunk) {
        v16h a, bm;
#pragma unroll
        for (int v = 0; v < 8; ++v) {
#pragma unroll
            for (int hlf = 0; hlf < 2; ++hlf) {
                // ---- A fragment: ISA 16-bit A 16x32 layout ----
                const int kA = chunk * 32 + laneHi * 8 +
                               ((v >= 4) ? 16 : 0) + (v & 3) * 2 + hlf;
                const int kk = (kA < KTAPS) ? kA : (KTAPS - 1);   // clamp
                const int ci = (kk >= 49) ? 1 : 0;
                const int r  = kk - 49 * ci;
                const int kh = r / 7;
                const int kw = r - kh * 7;
                // LDS col (wwM + kw) holds tensor col (wwM + kw - 3); halo = 0
                const float av = tile[ci * LCH + kh * LPITCH + wwM + kw];
                a[2 * v + hlf] = (_Float16)((kA < KTAPS) ? av : 0.0f);

                // ---- B fragment: 16-bit B 32x16 layout, broadcast over N ----
                const int kB = chunk * 32 + laneHi * 8 +
                               ((v >= 4) ? 16 : 0) + (v & 3) * 2 + hlf;
                const int kc = (kB < KTAPS) ? kB : (KTAPS - 1);
                const float wvv = wts[kc];
                bm[2 * v + hlf] = (_Float16)((kB < KTAPS) ? wvv : 0.0f);
            }
        }
        acc = __builtin_amdgcn_wmma_f32_16x16x32_f16(
                  false, a, false, bm, (short)0, acc, false, false);
    }

    // C/D layout: lanes 0-15 -> M=vgpr, lanes 16-31 -> M=vgpr+8; all N equal.
    if (M == 0) {
        const int base = b * HW_ + h * Wdim + wv * 16 + laneHi * 8;
#pragma unroll
        for (int v2 = 0; v2 < 8; ++v2) {
            y[base + v2] = 1.0f / (1.0f + __expf(-acc[v2]));
        }
    }
}

// ---------------------------------------------------------------------------
// Kernel 3: out = x * (1 + y).  x re-read (L2-resident from kernel 1);
// output written with nontemporal stores so it doesn't evict x from L2.
// ---------------------------------------------------------------------------
__global__ void sa_apply(const float* __restrict__ x,
                         const float* __restrict__ y,
                         float* __restrict__ out) {
    const size_t t   = (size_t)blockIdx.x * blockDim.x + threadIdx.x; // f4 idx
    const int    b   = (int)(t >> 18);      // 262144 f4 per image
    const int    hw4 = (int)(t & 1023);     // f4 index within spatial plane

    const f4 xv = ((const f4*)x)[t];
    const f4 yv = ((const f4*)y)[(size_t)b * 1024 + hw4];

    f4 o;
    o.x = xv.x * (1.0f + yv.x);
    o.y = xv.y * (1.0f + yv.y);
    o.z = xv.z * (1.0f + yv.z);
    o.w = xv.w * (1.0f + yv.w);
    __builtin_nontemporal_store(o, (f4*)out + t);
}

// ---------------------------------------------------------------------------
extern "C" void kernel_launch(void* const* d_in, const int* in_sizes, int n_in,
                              void* d_out, int out_size, void* d_ws, size_t ws_size,
                              hipStream_t stream) {
    const float* x = (const float*)d_in[0];   // [32,256,64,64]
    const float* W = (const float*)d_in[1];   // [1,2,7,7] = 98 floats
    float* out = (float*)d_out;

    float* avg = (float*)d_ws;                // 131072 floats
    float* mx  = avg + NPIX;                  // 131072 floats
    float* y   = mx + NPIX;                   // 131072 floats

    // Kernel 1: 32768 threads (one per plane-float4), 256 channels each.
    sa_reduce<<<128, 256, 0, stream>>>(x, avg, mx);

    // Kernel 2: one block (4 waves) per output row -> 32*64 = 2048 blocks.
    sa_conv_wmma<<<2048, 128, 0, stream>>>(avg, mx, W, y);

    // Kernel 3: 8388608 float4 elements -> 32768 blocks of 256.
    sa_apply<<<32768, 256, 0, stream>>>(x, y, out);
}
